// WindowAttention_71296457114375
// MI455X (gfx1250) — matrix operand order repaired and checked by reading, not compiled, not run on hardware
//
#include <hip/hip_runtime.h>

typedef __attribute__((ext_vector_type(16))) __bf16 v16bf;
typedef __attribute__((ext_vector_type(8)))  float  v8f;

union Frag {
    v16bf v;
    unsigned int u[8];
};

__device__ __forceinline__ unsigned short f2bf(float x) {
    unsigned int u = __float_as_uint(x);
    u += 0x7FFFu + ((u >> 16) & 1u);           // round-to-nearest-even
    return (unsigned short)(u >> 16);
}

#define DIMC  512
#define NTOK  64
#define NHEAD 16

// ---- LDS byte offsets (320 KB total per workgroup) ----
#define OFF_X   0u        // X bf16 [64][512]   (reused: attn heads 0..7)
#define OFF_Q   65536u    // Q bf16 [64][512]   (reused: xout bf16 [64][512])
#define OFF_K   131072u   // K bf16 [64][512]
#define OFF_VT  196608u   // V^T bf16 [512][64]
#define OFF_AH  262144u   // attn heads 8..15
#define LDS_BYTES 327680

// A-fragment K offset inside a 32-chunk (16-bit A 16x32 layout)
__device__ __forceinline__ int koffA(int j, int hi) {
    return ((j >> 2) << 4) + (hi << 3) + ((j & 3) << 1);
}
// B-fragment K offset inside a 32-chunk (16-bit B 32x16 layout)
__device__ __forceinline__ int koffB(int j, int hi) {
    return ((j >> 2) << 4) + (hi << 3) + ((j & 3) << 1);
}

// ---------------- prep: bf16-transpose weights + expand relative bias ----------------
__global__ void winattn_prep(const float* __restrict__ wq,
                             const float* __restrict__ wkv,
                             const float* __restrict__ wproj,
                             const float* __restrict__ btab,
                             const int*   __restrict__ rpi,
                             unsigned short* __restrict__ wqT,
                             unsigned short* __restrict__ wkvT,
                             unsigned short* __restrict__ wprojT,
                             float* __restrict__ biasf) {
    int i = blockIdx.x * blockDim.x + threadIdx.x;
    if (i < 512 * 512) {                       // wqT[n][k] = wq[k][n]
        int n = i >> 9, k = i & 511;
        wqT[i] = f2bf(wq[k * 512 + n]);
        return;
    }
    int j = i - 262144;
    if (j < 1024 * 512) {                      // wkvT[n][k] = wkv[k][n]
        int n = j >> 9, k = j & 511;
        wkvT[j] = f2bf(wkv[k * 1024 + n]);
        return;
    }
    int p = j - 524288;
    if (p < 512 * 512) {                       // wprojT[n][k] = wproj[k][n]
        int n = p >> 9, k = p & 511;
        wprojT[p] = f2bf(wproj[k * 512 + n]);
        return;
    }
    int t = p - 262144;
    if (t < NHEAD * 4096) {                    // biasf[h][m][n]
        int h = t >> 12, mn = t & 4095;
        biasf[t] = btab[rpi[mn] * NHEAD + h];
    }
}

// ---------------- fused window attention: 1 workgroup per window ----------------
__global__ void __launch_bounds__(512)
winattn_main(const float* __restrict__ q,
             const float* __restrict__ bq,
             const float* __restrict__ bkv,
             const float* __restrict__ bproj,
             const unsigned short* __restrict__ wqT,
             const unsigned short* __restrict__ wkvT,
             const unsigned short* __restrict__ wprojT,
             const float* __restrict__ biasf,
             float* __restrict__ out) {
    extern __shared__ __align__(16) unsigned char smem[];
    const int tid  = threadIdx.x;
    const int lane = tid & 31;
    const int w    = tid >> 5;      // wave id 0..15 == head id
    const int lo   = lane & 15;
    const int hi   = lane >> 4;
    const int b    = blockIdx.x;
    const float scale = 0.17677669529663687f;  // 1/sqrt(32)

    // ---- Phase 0: X = q[b] -> bf16 LDS ----
    const float* qb = q + (size_t)b * (NTOK * DIMC);
    for (int i = tid; i < NTOK * DIMC; i += 512)
        *(unsigned short*)(smem + OFF_X + 2u * i) = f2bf(qb[i]);
    __syncthreads();

    // ---- Phase 1: [Q|K|V] = X @ [wq|wkv] + bias ; wave w owns 6 column-tiles ----
    for (int mt = 0; mt < 4; ++mt) {
        v8f acc[6];
        #pragma unroll
        for (int i = 0; i < 6; ++i)
            #pragma unroll
            for (int r = 0; r < 8; ++r) acc[i][r] = 0.0f;

        for (int kk = 0; kk < 16; ++kk) {
            Frag A;
            const int row = 16 * mt + lo;
            #pragma unroll
            for (int j = 0; j < 8; ++j) {
                int k = 32 * kk + koffA(j, hi);
                A.u[j] = *(const unsigned int*)(smem + OFF_X + 2u * (row * DIMC + k));
            }
            #pragma unroll
            for (int i = 0; i < 6; ++i) {
                int colg = 16 * (6 * w + i) + lo;
                const unsigned short* wT = (colg < DIMC) ? (wqT + colg * DIMC)
                                                         : (wkvT + (colg - DIMC) * DIMC);
                Frag B;
                #pragma unroll
                for (int j = 0; j < 8; ++j) {
                    int k = 32 * kk + koffB(j, hi);
                    B.u[j] = *(const unsigned int*)(wT + k);
                }
                acc[i] = __builtin_amdgcn_wmma_f32_16x16x32_bf16(
                    false, A.v, false, B.v, (short)0, acc[i], false, false);
            }
        }
        #pragma unroll
        for (int i = 0; i < 6; ++i) {
            int colg = 16 * (6 * w + i) + lo;
            float bias = (colg < DIMC) ? bq[colg] : bkv[colg - DIMC];
            #pragma unroll
            for (int r = 0; r < 8; ++r) {
                int m = 16 * mt + r + 8 * hi;
                float val = acc[i][r] + bias;
                if (colg < DIMC) {
                    *(unsigned short*)(smem + OFF_Q + 2u * (m * DIMC + colg)) = f2bf(val * scale);
                } else if (colg < 2 * DIMC) {
                    *(unsigned short*)(smem + OFF_K + 2u * (m * DIMC + (colg - DIMC))) = f2bf(val);
                } else {
                    *(unsigned short*)(smem + OFF_VT + 2u * ((colg - 2 * DIMC) * NTOK + m)) = f2bf(val);
                }
            }
        }
    }
    __syncthreads();

    // ---- Phase 2: attn = softmax(Q_h K_h^T + bias) ; one wave per head ----
    const int h = w;
    const unsigned attn_base = (h < 8) ? (unsigned)(h * 8192)
                                       : (OFF_AH + (unsigned)(h - 8) * 8192);
    Frag Bk[4];
    #pragma unroll
    for (int nt = 0; nt < 4; ++nt) {
        int ntok = 16 * nt + lo;
        #pragma unroll
        for (int j = 0; j < 8; ++j) {
            int k = koffB(j, hi);
            Bk[nt].u[j] = *(const unsigned int*)(smem + OFF_K + 2u * (ntok * DIMC + 32 * h + k));
        }
    }
    float* out_attn = out + (size_t)33554432 + (((size_t)b * NHEAD + h) << 12);
    for (int mt = 0; mt < 4; ++mt) {
        Frag A;
        #pragma unroll
        for (int j = 0; j < 8; ++j) {
            int k = koffA(j, hi);
            A.u[j] = *(const unsigned int*)(smem + OFF_Q + 2u * ((16 * mt + lo) * DIMC + 32 * h + k));
        }
        v8f acc[4];
        #pragma unroll
        for (int nt = 0; nt < 4; ++nt) {
            #pragma unroll
            for (int r = 0; r < 8; ++r) acc[nt][r] = 0.0f;
            acc[nt] = __builtin_amdgcn_wmma_f32_16x16x32_bf16(
                false, A.v, false, Bk[nt].v, (short)0, acc[nt], false, false);
        }
        #pragma unroll
        for (int nt = 0; nt < 4; ++nt)
            #pragma unroll
            for (int r = 0; r < 8; ++r) {
                int m = 16 * mt + r + 8 * hi;
                int n = 16 * nt + lo;
                acc[nt][r] += biasf[(h * 64 + m) * 64 + n];
            }
        #pragma unroll
        for (int r = 0; r < 8; ++r) {   // softmax over n for the 2 rows this r covers
            float rmax = fmaxf(fmaxf(acc[0][r], acc[1][r]), fmaxf(acc[2][r], acc[3][r]));
            #pragma unroll
            for (int s = 1; s < 16; s <<= 1) rmax = fmaxf(rmax, __shfl_xor(rmax, s, 16));
            float sum = 0.0f;
            #pragma unroll
            for (int nt = 0; nt < 4; ++nt) {
                float e = __expf(acc[nt][r] - rmax);
                acc[nt][r] = e;
                sum += e;
            }
            #pragma unroll
            for (int s = 1; s < 16; s <<= 1) sum += __shfl_xor(sum, s, 16);
            float inv = 1.0f / sum;
            #pragma unroll
            for (int nt = 0; nt < 4; ++nt) acc[nt][r] *= inv;
        }
        #pragma unroll
        for (int nt = 0; nt < 4; ++nt)
            #pragma unroll
            for (int r = 0; r < 8; ++r) {
                int m = 16 * mt + r + 8 * hi;
                int n = 16 * nt + lo;
                out_attn[m * 64 + n] = acc[nt][r];
                *(unsigned short*)(smem + attn_base + 2u * (m * 64 + n)) = f2bf(acc[nt][r]);
            }
    }
    __syncthreads();   // all Q/K reads done -> Q region reusable as xout

    // ---- Phase 3: out_h = attn @ V_h ; bf16 result staged into xout (Q region) ----
    Frag Bv[2][2];
    #pragma unroll
    for (int n2 = 0; n2 < 2; ++n2)
        #pragma unroll
        for (int kc = 0; kc < 2; ++kc) {
            int vrow = 32 * h + 16 * n2 + lo;
            #pragma unroll
            for (int j = 0; j < 8; ++j) {
                int k = 32 * kc + koffB(j, hi);
                Bv[n2][kc].u[j] = *(const unsigned int*)(smem + OFF_VT + 2u * (vrow * NTOK + k));
            }
        }
    for (int mt = 0; mt < 4; ++mt) {
        v8f acc2[2];
        #pragma unroll
        for (int i = 0; i < 2; ++i)
            #pragma unroll
            for (int r = 0; r < 8; ++r) acc2[i][r] = 0.0f;
        #pragma unroll
        for (int kc = 0; kc < 2; ++kc) {
            Frag A;
            #pragma unroll
            for (int j = 0; j < 8; ++j) {
                int k = 32 * kc + koffA(j, hi);
                A.u[j] = *(const unsigned int*)(smem + attn_base + 2u * ((16 * mt + lo) * 64 + k));
            }
            #pragma unroll
            for (int n2 = 0; n2 < 2; ++n2)
                acc2[n2] = __builtin_amdgcn_wmma_f32_16x16x32_bf16(
                    false, A.v, false, Bv[n2][kc].v, (short)0, acc2[n2], false, false);
        }
        #pragma unroll
        for (int n2 = 0; n2 < 2; ++n2)
            #pragma unroll
            for (int r = 0; r < 8; ++r) {
                int m = 16 * mt + r + 8 * hi;
                int col = 32 * h + 16 * n2 + lo;
                *(unsigned short*)(smem + OFF_Q + 2u * (m * DIMC + col)) = f2bf(acc2[n2][r]);
            }
    }
    __syncthreads();

    // ---- Phase 4: x = xout @ wproj + bproj ; wave w owns 2 column-tiles ----
    for (int mt = 0; mt < 4; ++mt) {
        v8f acc2[2];
        #pragma unroll
        for (int i = 0; i < 2; ++i)
            #pragma unroll
            for (int r = 0; r < 8; ++r) acc2[i][r] = 0.0f;
        for (int kk = 0; kk < 16; ++kk) {
            Frag A;
            #pragma unroll
            for (int j = 0; j < 8; ++j) {
                int k = 32 * kk + koffA(j, hi);
                A.u[j] = *(const unsigned int*)(smem + OFF_Q + 2u * ((16 * mt + lo) * DIMC + k));
            }
            #pragma unroll
            for (int i = 0; i < 2; ++i) {
                int colg = 16 * (2 * w + i) + lo;
                Frag B;
                #pragma unroll
                for (int j = 0; j < 8; ++j) {
                    int k = 32 * kk + koffB(j, hi);
                    B.u[j] = *(const unsigned int*)(wprojT + colg * DIMC + k);
                }
                acc2[i] = __builtin_amdgcn_wmma_f32_16x16x32_bf16(
                    false, A.v, false, B.v, (short)0, acc2[i], false, false);
            }
        }
        #pragma unroll
        for (int i = 0; i < 2; ++i) {
            int colg = 16 * (2 * w + i) + lo;
            float bp = bproj[colg];
            #pragma unroll
            for (int r = 0; r < 8; ++r) {
                int m = 16 * mt + r + 8 * hi;
                out[((size_t)b * NTOK + m) * DIMC + colg] = acc2[i][r] + bp;
            }
        }
    }
}

extern "C" void kernel_launch(void* const* d_in, const int* in_sizes, int n_in,
                              void* d_out, int out_size, void* d_ws, size_t ws_size,
                              hipStream_t stream) {
    const float* q     = (const float*)d_in[0];
    const float* wq    = (const float*)d_in[1];
    const float* bq    = (const float*)d_in[2];
    const float* wkv   = (const float*)d_in[3];
    const float* bkv   = (const float*)d_in[4];
    const float* wproj = (const float*)d_in[5];
    const float* bproj = (const float*)d_in[6];
    const float* btab  = (const float*)d_in[7];
    const int*   rpi   = (const int*)d_in[8];

    // workspace layout (2.25 MB)
    unsigned short* wqT    = (unsigned short*)d_ws;
    unsigned short* wkvT   = (unsigned short*)((char*)d_ws + 524288);
    unsigned short* wprojT = (unsigned short*)((char*)d_ws + 1572864);
    float*          biasf  = (float*)((char*)d_ws + 2097152);

    winattn_prep<<<4352, 256, 0, stream>>>(wq, wkv, wproj, btab, rpi,
                                           wqT, wkvT, wprojT, biasf);

    (void)hipFuncSetAttribute(reinterpret_cast<const void*>(&winattn_main),
                              hipFuncAttributeMaxDynamicSharedMemorySize, LDS_BYTES);
    winattn_main<<<1024, 512, LDS_BYTES, stream>>>(q, bq, bkv, bproj,
                                                   wqT, wkvT, wprojT, biasf,
                                                   (float*)d_out);
}
